// GNN_79611513798758
// MI455X (gfx1250) — matrix operand order repaired and checked by reading, not compiled, not run on hardware
//
#include <hip/hip_runtime.h>

#define BS 8
#define NN 256
#define FD 16
#define HID 128

typedef __attribute__((ext_vector_type(16))) __bf16 v16bf;
typedef __attribute__((ext_vector_type(8)))  float  v8f;

__device__ __forceinline__ unsigned f2bf_rne(float f) {
  unsigned u = __float_as_uint(f);
  unsigned r = u + 0x7FFFu + ((u >> 16) & 1u);   // round-to-nearest-even bf16
  return r >> 16;
}
__device__ __forceinline__ unsigned pack2bf(float lo, float hi) {
  return (f2bf_rne(lo) & 0xFFFFu) | (f2bf_rne(hi) << 16);
}

// ---------------------------------------------------------------------------
// P1: per-node halves of edge layer 1.
// A[node][h]  = s4 . W1[0:4 , h]
// Bv[node][h] = s4 . W1[4:8 , h] + b1[h]
// ---------------------------------------------------------------------------
__global__ __launch_bounds__(256) void gnn_prep_AB(
    const float* __restrict__ states, const float* __restrict__ W1,
    const float* __restrict__ b1, float* __restrict__ A, float* __restrict__ Bv) {
  int idx  = blockIdx.x * 256 + threadIdx.x;     // 0 .. 2048*128-1
  int node = idx >> 7;
  int h    = idx & 127;
  const float* s = states + node * FD;
  float a = 0.f, bb = b1[h];
#pragma unroll
  for (int f = 0; f < 4; ++f) {
    float sv = s[f];
    a  += sv * W1[f * HID + h];
    bb += sv * W1[(4 + f) * HID + h];
  }
  A[idx] = a;
  Bv[idx] = bb;
}

// ---------------------------------------------------------------------------
// P2: pack W2 (128x128, row-major [k][n]) into bf16 WMMA B-operand layout.
// Slot [((kt*8+nt)*32+lane)*8+v] = {W2[k0][n], W2[k0+1][n]},
//   n  = nt*16 + (lane&15),  k0 = kt*32 + (lane>=16 ? 16:0) + 2v
// ---------------------------------------------------------------------------
__global__ __launch_bounds__(256) void gnn_pack_W2(
    const float* __restrict__ W2, unsigned* __restrict__ Wp) {
  int t    = blockIdx.x * 256 + threadIdx.x;     // 0..8191
  int v    = t & 7;
  int lane = (t >> 3) & 31;
  int nt   = (t >> 8) & 7;
  int kt   = t >> 11;
  int n    = nt * 16 + (lane & 15);
  int k0   = kt * 32 + ((lane >> 4) & 1) * 16 + 2 * v;
  Wp[t] = pack2bf(W2[k0 * HID + n], W2[(k0 + 1) * HID + n]);
}

// ---------------------------------------------------------------------------
// P3: dominant kernel. One workgroup per (b,i). 8 waves, wave wv owns output
// cols [wv*16, wv*16+16). W2 operand hoisted into 32 registers per wave.
// Double-buffered x1 tile: build chunk jc+1 while WMMAs consume chunk jc;
// one workgroup barrier per iteration.
// ---------------------------------------------------------------------------
__global__ __launch_bounds__(256) void gnn_edge_wmma(
    const float* __restrict__ A, const float* __restrict__ Bv,
    const unsigned* __restrict__ Wp, const float* __restrict__ b2,
    float* __restrict__ inter) {
  __shared__ __align__(16) unsigned wlds[8192];     // 32 KB packed W2
  __shared__ __align__(16) unsigned x1p[2][1024];   // 2 x 4 KB x1 tiles
  __shared__ float As[HID];                         // A row for this i

  const int tid  = threadIdx.x;
  const int node = blockIdx.x;      // b*N + i
  const int b    = node >> 8;

  // stage packed W2 into LDS with b128 transfers
  {
    const uint4* g4 = (const uint4*)Wp;
    uint4* l4 = (uint4*)wlds;
    for (int t = tid; t < 2048; t += 256) l4[t] = g4[t];
  }
  if (tid < HID) As[tid] = A[(size_t)node * HID + tid];

  const int lane = tid & 31;
  const int wv   = tid >> 5;                          // col tile (nt)
  const float b2c = b2[wv * 16 + (lane & 15)];

  // x1-builder indices: tid -> (kt, lane', 4 dwords)
  const int bkt   = tid >> 6;                         // 0..3
  const int blp   = tid & 31;                         // operand lane
  const int bhalf = (tid >> 5) & 1;                   // which 4 dwords
  const int bm    = blp & 15;                         // j row in chunk
  const int kbase = bkt * 32 + ((blp >> 4) & 1) * 16; // contiguous k = 16h + s

  union U { unsigned u[8]; v16bf v; };

  auto build = [&](int jc, int buf) {
    const float* bvrow = Bv + (size_t)(b * NN + jc * 16 + bm) * HID;
#pragma unroll
    for (int vv = 0; vv < 4; ++vv) {
      int v   = bhalf * 4 + vv;
      int off = 2 * v;                                // K pairs 0..15 contiguous
      int k0  = kbase + off;
      float x0 = fmaxf(As[k0]     + bvrow[k0],     0.f);
      float x1 = fmaxf(As[k0 + 1] + bvrow[k0 + 1], 0.f);
      x1p[buf][(bkt * 32 + blp) * 8 + v] = pack2bf(x0, x1);
    }
  };

  __syncthreads();

  // hoist loop-invariant W2 B-operands into registers (4 kt x 8 VGPRs)
  U wop[4];
#pragma unroll
  for (int kt = 0; kt < 4; ++kt) {
    const unsigned* wp = &wlds[((kt * 8 + wv) * 32 + lane) * 8];
#pragma unroll
    for (int q = 0; q < 8; ++q) wop[kt].u[q] = wp[q];
  }

  build(0, 0);
  __syncthreads();

  float acc = 0.f;
  for (int jc = 0; jc < 16; ++jc) {
    const int cur = jc & 1;
    if (jc < 15) build(jc + 1, cur ^ 1);   // overlap with this chunk's WMMAs

    v8f c = {};
#pragma unroll
    for (int kt = 0; kt < 4; ++kt) {
      U aop;
      const unsigned* ap = &x1p[cur][(kt * 32 + lane) * 8];
#pragma unroll
      for (int q = 0; q < 8; ++q) aop.u[q] = ap[q];
      c = __builtin_amdgcn_wmma_f32_16x16x32_bf16(false, aop.v, false, wop[kt].v,
                                                  (short)0, c, false, false);
    }
    // epilogue: relu(D + b2) then sum over the 16 j rows of this chunk
#pragma unroll
    for (int r = 0; r < 8; ++r) acc += fmaxf(c[r] + b2c, 0.f);
    __syncthreads();
  }

  acc += __shfl_xor(acc, 16, 32);   // merge row halves (lanes L <-> L+16)
  if (lane < 16)
    inter[(size_t)node * HID + wv * 16 + lane] = acc;
}

// ---------------------------------------------------------------------------
// P4: node MLP (144->128 relu, 128->128 relu, 128->16). Tiny; one block/node.
// ---------------------------------------------------------------------------
__global__ __launch_bounds__(128) void gnn_node_mlp(
    const float* __restrict__ states, const float* __restrict__ inter,
    const float* __restrict__ Wn0, const float* __restrict__ bn0,
    const float* __restrict__ Wn1, const float* __restrict__ bn1,
    const float* __restrict__ Wn2, const float* __restrict__ bn2,
    float* __restrict__ out) {
  __shared__ float y0[FD + HID];
  __shared__ float h1[HID];
  __shared__ float h2[HID];
  const int t = threadIdx.x;
  const int node = blockIdx.x;
  if (t < FD) y0[t] = states[node * FD + t];
  y0[FD + t] = inter[(size_t)node * HID + t];
  __syncthreads();
  float s = bn0[t];
  for (int k = 0; k < FD + HID; ++k) s += y0[k] * Wn0[k * HID + t];
  h1[t] = fmaxf(s, 0.f);
  __syncthreads();
  s = bn1[t];
  for (int k = 0; k < HID; ++k) s += h1[k] * Wn1[k * HID + t];
  h2[t] = fmaxf(s, 0.f);
  __syncthreads();
  if (t < FD) {
    s = bn2[t];
    for (int k = 0; k < HID; ++k) s += h2[k] * Wn2[k * FD + t];
    out[node * FD + t] = s;
  }
}

extern "C" void kernel_launch(void* const* d_in, const int* in_sizes, int n_in,
                              void* d_out, int out_size, void* d_ws, size_t ws_size,
                              hipStream_t stream) {
  (void)in_sizes; (void)n_in; (void)out_size; (void)ws_size;
  const float* states = (const float*)d_in[0];
  const float* W1  = (const float*)d_in[1];   // edge_Ws[0]  (8,128)
  const float* W2  = (const float*)d_in[2];   // edge_Ws[1]  (128,128)
  const float* b1  = (const float*)d_in[3];   // edge_bs[0]
  const float* b2  = (const float*)d_in[4];   // edge_bs[1]
  const float* Wn0 = (const float*)d_in[5];   // node_Ws[0] (144,128)
  const float* Wn1 = (const float*)d_in[6];   // node_Ws[1] (128,128)
  const float* Wn2 = (const float*)d_in[7];   // node_Ws[2] (128,16)
  const float* bn0 = (const float*)d_in[8];
  const float* bn1 = (const float*)d_in[9];
  const float* bn2 = (const float*)d_in[10];

  float* ws    = (float*)d_ws;
  float* A     = ws;                          // 2048*128 f32 = 1 MB
  float* Bv    = ws + 262144;                 // 1 MB
  float* inter = ws + 524288;                 // 1 MB
  unsigned* Wp = (unsigned*)(ws + 786432);    // 32 KB packed W2

  gnn_prep_AB <<<1024, 256, 0, stream>>>(states, W1, b1, A, Bv);
  gnn_pack_W2 <<<32,   256, 0, stream>>>(W2, Wp);
  gnn_edge_wmma<<<BS * NN, 256, 0, stream>>>(A, Bv, Wp, b2, inter);
  gnn_node_mlp<<<BS * NN, 128, 0, stream>>>(states, inter, Wn0, bn0, Wn1, bn1,
                                            Wn2, bn2, (float*)d_out);
}
